// SelfAttentionV2_3685081940288
// MI455X (gfx1250) — matrix operand order repaired and checked by reading, compiled, not run on hardware
//
#include <hip/hip_runtime.h>
#include <hip/hip_bf16.h>

typedef __attribute__((ext_vector_type(16))) _Float16 v16h;
typedef __attribute__((ext_vector_type(8)))  _Float16 v8h;
typedef __attribute__((ext_vector_type(8)))  float    v8f;
typedef int v4i __attribute__((vector_size(4 * sizeof(int))));

#define HIDDEN 1024
#define BATCH  4
#define SEQ    4096
#define NOUT   (3 * HIDDEN)          // 3072
#define MROWS  (BATCH * SEQ)         // 16384

// LDS slab: 128 rows x 32 halves, padded to 40 halves (80B row stride -> 20
// dword stride, gcd(20,64)=4, 16-row fragment reads hit 16 distinct banks).
#define LDS_LD 40

#if defined(__has_builtin)
#if __has_builtin(__builtin_amdgcn_global_load_async_to_lds_b128)
#define HAVE_ASYNC_LDS 1
#endif
#endif
#ifndef HAVE_ASYNC_LDS
#define HAVE_ASYNC_LDS 0
#endif

// ---------------------------------------------------------------------------
// Async global->LDS copy of 16B (ASYNCcnt-tracked), with sync fallback
// ---------------------------------------------------------------------------
__device__ __forceinline__ void cp_b128(const _Float16* g, _Float16* l) {
#if HAVE_ASYNC_LDS
    __builtin_amdgcn_global_load_async_to_lds_b128(
        (__attribute__((address_space(1))) v4i*)g,
        (__attribute__((address_space(3))) v4i*)l,
        0, 0);
#else
    *(v8h*)l = *(const v8h*)g;     // global_load_b128 + ds_store_b128
#endif
}

#if HAVE_ASYNC_LDS
__device__ __forceinline__ void wait_async_le4() {
#if __has_builtin(__builtin_amdgcn_s_wait_asynccnt)
    __builtin_amdgcn_s_wait_asynccnt(4);
#else
    asm volatile("s_wait_asynccnt 0x4" ::: "memory");
#endif
}
__device__ __forceinline__ void wait_async_0() {
#if __has_builtin(__builtin_amdgcn_s_wait_asynccnt)
    __builtin_amdgcn_s_wait_asynccnt(0);
#else
    asm volatile("s_wait_asynccnt 0x0" ::: "memory");
#endif
}
#endif

__device__ __forceinline__ v8f wmma16x16x32(v16h a, v16h b, v8f c) {
    return __builtin_amdgcn_wmma_f32_16x16x32_f16(
        /*neg_a=*/false, a, /*neg_b=*/false, b,
        /*c_mod=*/(short)0, c, /*reuse_a=*/false, /*reuse_b=*/false);
}

// ---------------------------------------------------------------------------
// LDS-staged, double-buffered GEMM core.
// A is [M, Kdim] row-major, Bm is [N, Kdim] row-major  =>  C = A * Bm^T.
// Block: 256 threads (8 waves), block tile 128x128.
// Wave (2x4 grid) computes a 64x32 subtile = 4x2 WMMA tiles, acc[4][2].
// Fragment layouts used here (wave32; A and B share one k order):
//   A lane L: row M=L%16; halves = 16 contiguous K at 16*(L/16)
//   B lane L: col N=L%16; halves = 16 contiguous K at 16*(L/16)
//   C lane L, vgpr v: (M = v + 8*(L/16), N = L%16)
// ---------------------------------------------------------------------------
__device__ __forceinline__ void gemm_core_staged(const _Float16* __restrict__ A, int lda,
                                                 const _Float16* __restrict__ Bm, int ldb,
                                                 int Kdim, int m0blk, int n0blk,
                                                 v8f acc[4][2]) {
    __shared__ _Float16 As[2][128][LDS_LD];
    __shared__ _Float16 Bs[2][128][LDS_LD];

    const int t    = threadIdx.x;
    const int lane = t & 31;
    const int wave = t >> 5;
    const int m0w  = (wave >> 2) * 64;   // 0 or 64
    const int n0w  = (wave & 3) * 32;    // 0,32,64,96

    // staging assignment: thread t copies 16 halves (2 x b128) per matrix
    const int srow = t >> 1;             // 0..127
    const int scol = (t & 1) * 16;       // 0 or 16
    const _Float16* gA = A  + (size_t)(m0blk + srow) * (size_t)lda + scol;
    const _Float16* gB = Bm + (size_t)(n0blk + srow) * (size_t)ldb + scol;

    // prologue: stage k-slab 0 into buffer 0
    cp_b128(gA,     &As[0][srow][scol]);
    cp_b128(gA + 8, &As[0][srow][scol + 8]);
    cp_b128(gB,     &Bs[0][srow][scol]);
    cp_b128(gB + 8, &Bs[0][srow][scol + 8]);

    for (int k0 = 0; k0 < Kdim; k0 += 32) {
        const int  buf  = (k0 >> 5) & 1;
        const bool more = (k0 + 32) < Kdim;

        if (more) {   // stage next k-slab into the other buffer (stays in flight)
            const _Float16* nA = gA + k0 + 32;
            const _Float16* nB = gB + k0 + 32;
            cp_b128(nA,     &As[buf ^ 1][srow][scol]);
            cp_b128(nA + 8, &As[buf ^ 1][srow][scol + 8]);
            cp_b128(nB,     &Bs[buf ^ 1][srow][scol]);
            cp_b128(nB + 8, &Bs[buf ^ 1][srow][scol + 8]);
        }
#if HAVE_ASYNC_LDS
        if (more) wait_async_le4();   // current slab's 4 copies complete, next 4 in flight
        else      wait_async_0();
#endif
        __syncthreads();

        v16h a[4], b[2];
#pragma unroll
        for (int i = 0; i < 4; ++i) {
            const _Float16* p = &As[buf][m0w + 16 * i + (lane & 15)][(lane >> 4) << 4];
            v8h lo = *(const v8h*)p;
            v8h hi = *(const v8h*)(p + 8);
            v16h f;
#pragma unroll
            for (int e = 0; e < 8; ++e) { f[e] = lo[e]; f[e + 8] = hi[e]; }
            a[i] = f;
        }
#pragma unroll
        for (int j = 0; j < 2; ++j) {
            const _Float16* p = &Bs[buf][n0w + 16 * j + (lane & 15)][(lane >> 4) << 4];
            b[j] = *(const v16h*)p;
        }
#pragma unroll
        for (int i = 0; i < 4; ++i)
#pragma unroll
            for (int j = 0; j < 2; ++j)
                acc[i][j] = wmma16x16x32(a[i], b[j], acc[i][j]);
        __syncthreads();   // everyone done reading buf before it is restaged
    }
}

// ---------------------------------------------------------------------------
// Kernel 1: fp32 -> f16 convert
// ---------------------------------------------------------------------------
__global__ void cvt_f32_f16(const float* __restrict__ in, _Float16* __restrict__ out, int n) {
    int i = blockIdx.x * blockDim.x + threadIdx.x;
    if (i < n) out[i] = (_Float16)in[i];
}

// ---------------------------------------------------------------------------
// Kernel 2: QKV = X * W^T + bias ; scatter into Q [b,s,h], K [b,s,h], Vt [b,h,s]
// ---------------------------------------------------------------------------
__global__ void __launch_bounds__(256)
qkv_gemm(const _Float16* __restrict__ Xh,   // [16384, 1024]
         const _Float16* __restrict__ Wh,   // [3072, 1024]
         const float* __restrict__ bias,    // [3072]
         _Float16* __restrict__ Qh, _Float16* __restrict__ Kh, _Float16* __restrict__ Vt) {
    const int lane = threadIdx.x & 31;
    const int wave = threadIdx.x >> 5;
    const int m0blk = blockIdx.y * 128;
    const int n0blk = blockIdx.x * 128;

    v8f acc[4][2] = {};
    gemm_core_staged(Xh, HIDDEN, Wh, HIDDEN, HIDDEN, m0blk, n0blk, acc);

    const int m0w = m0blk + (wave >> 2) * 64;
    const int n0w = n0blk + (wave & 3) * 32;
    const int mlane = (lane >> 4) << 3;
    const int nlane = lane & 15;
#pragma unroll
    for (int j = 0; j < 2; ++j) {
        const int n = n0w + 16 * j + nlane;
        const float bv = bias[n];
#pragma unroll
        for (int i = 0; i < 4; ++i) {
#pragma unroll
            for (int v = 0; v < 8; ++v) {
                const int m = m0w + 16 * i + mlane + v;
                const float val = acc[i][j][v] + bv;
                const int b = m >> 12;            // /SEQ
                const int s = m & (SEQ - 1);
                const size_t base = (size_t)b * SEQ * HIDDEN;
                if (n < HIDDEN)
                    Qh[base + (size_t)s * HIDDEN + n] = (_Float16)val;
                else if (n < 2 * HIDDEN)
                    Kh[base + (size_t)s * HIDDEN + (n - HIDDEN)] = (_Float16)val;
                else  // store V transposed: Vt[b][h][s]
                    Vt[base + (size_t)(n - 2 * HIDDEN) * SEQ + s] = (_Float16)val;
            }
        }
    }
}

// ---------------------------------------------------------------------------
// Kernel 3: S = (Q * K^T) / sqrt(d), per batch, f16 output
// ---------------------------------------------------------------------------
__global__ void __launch_bounds__(256)
scores_gemm(const _Float16* __restrict__ Qh, const _Float16* __restrict__ Kh,
            _Float16* __restrict__ Sbuf) {
    const int lane = threadIdx.x & 31;
    const int wave = threadIdx.x >> 5;
    const size_t boff = (size_t)blockIdx.z * SEQ * HIDDEN;
    const int m0blk = blockIdx.y * 128;
    const int n0blk = blockIdx.x * 128;

    v8f acc[4][2] = {};
    gemm_core_staged(Qh + boff, HIDDEN, Kh + boff, HIDDEN, HIDDEN, m0blk, n0blk, acc);

    _Float16* S = Sbuf + (size_t)blockIdx.z * SEQ * SEQ;
    const float scale = 0.03125f;   // 1/sqrt(1024)
    const int m0w = m0blk + (wave >> 2) * 64;
    const int n0w = n0blk + (wave & 3) * 32;
    const int mlane = (lane >> 4) << 3;
    const int nlane = lane & 15;
#pragma unroll
    for (int i = 0; i < 4; ++i)
#pragma unroll
        for (int j = 0; j < 2; ++j)
#pragma unroll
            for (int v = 0; v < 8; ++v) {
                const int m = m0w + 16 * i + mlane + v;
                const int n = n0w + 16 * j + nlane;
                S[(size_t)m * SEQ + n] = (_Float16)(acc[i][j][v] * scale);
            }
}

// ---------------------------------------------------------------------------
// Kernel 4: in-place row softmax over f16 rows of length SEQ (4096)
// ---------------------------------------------------------------------------
__global__ void __launch_bounds__(256)
softmax_rows(_Float16* __restrict__ Sbuf) {
    __shared__ float red[256];
    _Float16* rowp = Sbuf + (size_t)blockIdx.x * SEQ;
    const int t = threadIdx.x;

    float vals[16];
    float lmax = -3.0e38f;
#pragma unroll
    for (int i = 0; i < 16; ++i) {
        vals[i] = (float)rowp[t + i * 256];
        lmax = fmaxf(lmax, vals[i]);
    }
    red[t] = lmax;
    __syncthreads();
    for (int s = 128; s > 0; s >>= 1) {
        if (t < s) red[t] = fmaxf(red[t], red[t + s]);
        __syncthreads();
    }
    const float m = red[0];
    __syncthreads();

    float lsum = 0.f;
#pragma unroll
    for (int i = 0; i < 16; ++i) {
        vals[i] = __expf(vals[i] - m);
        lsum += vals[i];
    }
    red[t] = lsum;
    __syncthreads();
    for (int s = 128; s > 0; s >>= 1) {
        if (t < s) red[t] += red[t + s];
        __syncthreads();
    }
    const float inv = 1.0f / red[0];
#pragma unroll
    for (int i = 0; i < 16; ++i)
        rowp[t + i * 256] = (_Float16)(vals[i] * inv);
}

// ---------------------------------------------------------------------------
// Kernel 5: out = P * V  (= P * Vt^T with Vt [1024, SEQ]), f32 output
// ---------------------------------------------------------------------------
__global__ void __launch_bounds__(256)
out_gemm(const _Float16* __restrict__ Sbuf, const _Float16* __restrict__ Vt,
         float* __restrict__ out) {
    const int lane = threadIdx.x & 31;
    const int wave = threadIdx.x >> 5;
    const _Float16* A  = Sbuf + (size_t)blockIdx.z * SEQ * SEQ;     // [4096, 4096]
    const _Float16* Bm = Vt   + (size_t)blockIdx.z * HIDDEN * SEQ;  // [1024, 4096]
    const int m0blk = blockIdx.y * 128;
    const int n0blk = blockIdx.x * 128;

    v8f acc[4][2] = {};
    gemm_core_staged(A, SEQ, Bm, SEQ, SEQ, m0blk, n0blk, acc);

    float* O = out + (size_t)blockIdx.z * SEQ * HIDDEN;
    const int m0w = m0blk + (wave >> 2) * 64;
    const int n0w = n0blk + (wave & 3) * 32;
    const int mlane = (lane >> 4) << 3;
    const int nlane = lane & 15;
#pragma unroll
    for (int i = 0; i < 4; ++i)
#pragma unroll
        for (int j = 0; j < 2; ++j)
#pragma unroll
            for (int v = 0; v < 8; ++v) {
                const int m = m0w + 16 * i + mlane + v;
                const int n = n0w + 16 * j + nlane;
                O[(size_t)m * HIDDEN + n] = acc[i][j][v];
            }
}

// ---------------------------------------------------------------------------
// Launch
// ---------------------------------------------------------------------------
extern "C" void kernel_launch(void* const* d_in, const int* in_sizes, int n_in,
                              void* d_out, int out_size, void* d_ws, size_t ws_size,
                              hipStream_t stream) {
    const float* X    = (const float*)d_in[0];   // [4, 4096, 1024]
    const float* W    = (const float*)d_in[1];   // [3072, 1024]
    const float* bias = (const float*)d_in[2];   // [3072]
    float* out = (float*)d_out;                  // [4, 4096, 1024] f32

    // Workspace layout (f16 elements): total ~275 MB
    _Float16* Xh = (_Float16*)d_ws;                            // 16384*1024
    _Float16* Wh = Xh + (size_t)MROWS * HIDDEN;                //  3072*1024
    _Float16* Qh = Wh + (size_t)NOUT * HIDDEN;                 // 4*4096*1024
    _Float16* Kh = Qh + (size_t)BATCH * SEQ * HIDDEN;
    _Float16* Vt = Kh + (size_t)BATCH * SEQ * HIDDEN;          // transposed V
    _Float16* Sb = Vt + (size_t)BATCH * SEQ * HIDDEN;          // 4*4096*4096 scores

    const int nX = MROWS * HIDDEN;
    cvt_f32_f16<<<(nX + 255) / 256, 256, 0, stream>>>(X, Xh, nX);
    const int nW = NOUT * HIDDEN;
    cvt_f32_f16<<<(nW + 255) / 256, 256, 0, stream>>>(W, Wh, nW);

    qkv_gemm<<<dim3(NOUT / 128, MROWS / 128, 1), 256, 0, stream>>>(Xh, Wh, bias, Qh, Kh, Vt);
    scores_gemm<<<dim3(SEQ / 128, SEQ / 128, BATCH), 256, 0, stream>>>(Qh, Kh, Sb);
    softmax_rows<<<dim3(BATCH * SEQ), 256, 0, stream>>>(Sb);
    out_gemm<<<dim3(HIDDEN / 128, SEQ / 128, BATCH), 256, 0, stream>>>(Sb, Vt, out);
}
